// Seq2SeqWithAttention_21148418965937
// MI455X (gfx1250) — matrix operand order repaired and checked by reading, not compiled, not run on hardware
//
#include <hip/hip_runtime.h>

// ---------------------------------------------------------------------------
// Seq2Seq + Bahdanau attention for MI455X (gfx1250, wave32, WMMA).
// Strategy:
//  * Convert all weight matrices + gathered embeddings to bf16 once (L2-friendly;
//    Wo drops 262MB->131MB and fits the 192MB L2).
//  * Hoist all input projections into big bf16 WMMA GEMMs up front.
//  * Per-timestep kernels do only the recurrent GEMM (h@Whh^T [+ ctx@Wih_ctx^T])
//    fused with the LSTM cell; each wave owns the same 16-wide slice of all
//    four gates so the nonlinearity is register-resident.
//  * All 64 logits projections are deferred into ONE batched GEMM
//    Z[T*B,2048] @ Wo^T[2048,32000]; Wo streams from HBM once, and 4x4
//    register blocking (64x64 macro-tile per wave) cuts L2 traffic ~4x
//    (33GB -> 8.2GB) versus one-tile-per-wave.
// ---------------------------------------------------------------------------

typedef __attribute__((ext_vector_type(16))) __bf16 v16bf;
typedef __attribute__((ext_vector_type(8)))  __bf16 v8bf;
typedef __attribute__((ext_vector_type(8)))  float  v8f;

#define BB 32            // batch
#define SS 64            // src len
#define TT 64            // tgt len
#define EE 256           // embed dim
#define HH 512           // enc hidden per dir
#define DH 1024          // decoder hidden
#define AA 128           // attention dim
#define VT 32000         // vocab

__device__ __forceinline__ float sigm(float x) { return 1.0f / (1.0f + __expf(-x)); }

__device__ __forceinline__ __bf16 to_bf16(float f) {
  unsigned u = __builtin_bit_cast(unsigned, f);
  u += 0x7FFFu + ((u >> 16) & 1u);            // round-to-nearest-even
  unsigned short h = (unsigned short)(u >> 16);
  return __builtin_bit_cast(__bf16, h);
}
__device__ __forceinline__ float bf2f(__bf16 b) {
  unsigned short h = __builtin_bit_cast(unsigned short, b);
  unsigned u = ((unsigned)h) << 16;
  return __builtin_bit_cast(float, u);
}

// A-fragment (16x32 bf16, M rows striped over lanes; K split 8+8 per lane half)
__device__ __forceinline__ v16bf load_afrag(const __bf16* __restrict__ base, int lda,
                                            int m0, int k0) {
  int lane = threadIdx.x & 31;
  int half = lane >> 4;
  const __bf16* p = base + (long long)(m0 + (lane & 15)) * lda + k0 + (half << 3);
  v8bf lo = *(const v8bf*)p;          // K = k0+8h .. +7
  v8bf hi = *(const v8bf*)(p + 16);   // K = k0+16+8h .. +7
  v16bf r;
#pragma unroll
  for (int i = 0; i < 8; ++i) { r[i] = lo[i]; r[i + 8] = hi[i]; }
  return r;
}

// B-fragment (32x16 bf16) built directly from row-major W[N,K] -> computes A@W^T.
// lane = N column, K split 8+8 per lane half (k0+8h .. +7, k0+16+8h .. +7).
__device__ __forceinline__ v16bf load_bfrag(const __bf16* __restrict__ w, int ldw,
                                            int n0, int k0) {
  int lane = threadIdx.x & 31;
  int half = lane >> 4;
  const __bf16* p = w + (long long)(n0 + (lane & 15)) * ldw + k0 + (half << 3);
  v8bf lo = *(const v8bf*)p;
  v8bf hi = *(const v8bf*)(p + 16);
  v16bf r;
#pragma unroll
  for (int i = 0; i < 8; ++i) { r[i] = lo[i]; r[i + 8] = hi[i]; }
  return r;
}

__device__ __forceinline__ v8f wmma_bf16(v16bf a, v16bf b, v8f c) {
  return __builtin_amdgcn_wmma_f32_16x16x32_bf16(false, a, false, b, (short)0, c,
                                                 false, false);
}

// ---------------------------------------------------------------------------
// Utility kernels
// ---------------------------------------------------------------------------
__global__ void k_cvt(const float* __restrict__ src, __bf16* __restrict__ dst,
                      long long n) {
  long long i = (long long)blockIdx.x * blockDim.x + threadIdx.x;
  long long stride = (long long)gridDim.x * blockDim.x;
  for (; i < n; i += stride) dst[i] = to_bf16(src[i]);
}
__global__ void k_zero_f32(float* p, int n) {
  int i = blockIdx.x * blockDim.x + threadIdx.x;
  if (i < n) p[i] = 0.0f;
}
__global__ void k_zero_bf16(__bf16* p, int n) {
  int i = blockIdx.x * blockDim.x + threadIdx.x;
  if (i < n) p[i] = to_bf16(0.0f);
}

// Embedding gather -> bf16, layout [L, B, E] (row = s*B + b), ids are [B, L].
__global__ void k_embed(const int* __restrict__ ids, const float* __restrict__ table,
                        __bf16* __restrict__ out, int L) {
  int row = blockIdx.x;             // s*B + b
  int s = row >> 5;                 // B == 32
  int b = row & 31;
  int id = ids[b * L + s];
  out[(long long)row * EE + threadIdx.x] =
      to_bf16(table[(long long)id * EE + threadIdx.x]);
}

// ---------------------------------------------------------------------------
// Generic GEMM: D[M,N] (f32) = A[M,K](bf16) @ W[N,K]^T(bf16) + bias.
// One wave per 16x16 tile; block = 128 threads = 4 waves.
// ---------------------------------------------------------------------------
__global__ void k_gemm(const __bf16* __restrict__ A, int lda,
                       const __bf16* __restrict__ W, int ldw,
                       const float* __restrict__ bias,
                       float* __restrict__ D, long long ldd,
                       int M, int N, int K) {
  int wave = (blockIdx.x * blockDim.x + threadIdx.x) >> 5;
  int ntiles = N >> 4;
  if (wave >= (M >> 4) * ntiles) return;
  int mt = wave / ntiles, nt = wave % ntiles;
  int lane = threadIdx.x & 31;
  v8f acc = {};
  for (int k = 0; k < K; k += 32) {
    v16bf a = load_afrag(A, lda, mt * 16, k);
    v16bf b = load_bfrag(W, ldw, nt * 16, k);
    acc = wmma_bf16(a, b, acc);
  }
  int nl = nt * 16 + (lane & 15);
  float bv = bias ? bias[nl] : 0.0f;
  int mb = mt * 16 + ((lane >> 4) << 3);
#pragma unroll
  for (int r = 0; r < 8; ++r)
    D[(long long)(mb + r) * ldd + nl] = acc[r] + bv;
}

// ---------------------------------------------------------------------------
// Batched logits GEMM: Z[T*B, 2048] @ Wo[VT,2048]^T + bo -> out[b, t, v].
// 4x4 register blocking: one wave owns a 64x64 macro-tile (16 WMMA tiles),
// so each A/B fragment load feeds 4 WMMAs. Wo (bf16, 131MB) is read 32x from
// L2 instead of 128x; Z (8MB) read 500x instead of 2000x.
// ---------------------------------------------------------------------------
__global__ void k_gemm_logits(const __bf16* __restrict__ Z, const __bf16* __restrict__ Wo,
                              const float* __restrict__ bo, float* __restrict__ out) {
  const int K = 2 * DH;            // 2048
  const int NG = (VT / 16) / 4;    // 500 groups of 64 columns
  const int MG = (TT * BB / 16) / 4;  // 32 groups of 64 rows
  int wave = (blockIdx.x * blockDim.x + threadIdx.x) >> 5;
  if (wave >= MG * NG) return;
  int mg = wave / NG, ng = wave % NG;
  int m0 = mg * 64, n0 = ng * 64;
  int lane = threadIdx.x & 31;
  v8f zero = {};
  v8f acc[4][4];
#pragma unroll
  for (int i = 0; i < 4; ++i)
#pragma unroll
    for (int j = 0; j < 4; ++j) acc[i][j] = zero;

  for (int k = 0; k < K; k += 32) {
    v16bf a[4], b[4];
#pragma unroll
    for (int i = 0; i < 4; ++i) a[i] = load_afrag(Z, K, m0 + 16 * i, k);
#pragma unroll
    for (int j = 0; j < 4; ++j) b[j] = load_bfrag(Wo, K, n0 + 16 * j, k);
#pragma unroll
    for (int i = 0; i < 4; ++i)
#pragma unroll
      for (int j = 0; j < 4; ++j) acc[i][j] = wmma_bf16(a[i], b[j], acc[i][j]);
  }

#pragma unroll
  for (int i = 0; i < 4; ++i) {
    int mb = m0 + 16 * i + ((lane >> 4) << 3);
#pragma unroll
    for (int j = 0; j < 4; ++j) {
      int nl = n0 + 16 * j + (lane & 15);
      float bb = bo[nl];
#pragma unroll
      for (int r = 0; r < 8; ++r) {
        int m = mb + r;                 // = t*B + b
        int bi = m & 31, ti = m >> 5;
        out[((long long)bi * TT + ti) * VT + nl] = acc[i][j][r] + bb;
      }
    }
  }
}

// ---------------------------------------------------------------------------
// Fused bidirectional encoder step: gates = gi[s] + h_prev@Whh^T + b; LSTM cell.
// 128 waves: dir = wave>>6; per dir: jt (H/16=32) x mt (2). Each wave owns the
// same 16-wide slice of all 4 gates -> cell epilogue in registers.
// grid 32 x 128 threads.
// ---------------------------------------------------------------------------
__global__ void k_enc_step_bi(const __bf16* __restrict__ hPrevF, int ldF,
                              const __bf16* __restrict__ hPrevB, int ldB,
                              const __bf16* __restrict__ WhhF,
                              const __bf16* __restrict__ WhhB,
                              const float* __restrict__ giF, const float* __restrict__ giB,
                              const float* __restrict__ bihF, const float* __restrict__ bhhF,
                              const float* __restrict__ bihB, const float* __restrict__ bhhB,
                              float* __restrict__ cF, float* __restrict__ cB,
                              __bf16* __restrict__ hOutF, __bf16* __restrict__ hOutB) {
  int gwave = (blockIdx.x * blockDim.x + threadIdx.x) >> 5;   // 0..127
  int dir = gwave >> 6;
  int w = gwave & 63;
  const __bf16* hPrev = dir ? hPrevB : hPrevF;
  int ldh = dir ? ldB : ldF;
  const __bf16* Whh = dir ? WhhB : WhhF;
  const float* gi = dir ? giB : giF;
  const float* bih = dir ? bihB : bihF;
  const float* bhh = dir ? bhhB : bhhF;
  float* c = dir ? cB : cF;
  __bf16* hOut = dir ? hOutB : hOutF;

  int lane = threadIdx.x & 31;
  int jt = w >> 1, mt = w & 1;
  int j0 = jt * 16, m0 = mt * 16;
  v8f ai = {}, af = {}, ag = {}, ao = {};
  for (int k = 0; k < HH; k += 32) {
    v16bf a = load_afrag(hPrev, ldh, m0, k);
    v16bf b0 = load_bfrag(Whh, HH, j0 + 0 * HH, k);
    v16bf b1 = load_bfrag(Whh, HH, j0 + 1 * HH, k);
    v16bf b2 = load_bfrag(Whh, HH, j0 + 2 * HH, k);
    v16bf b3 = load_bfrag(Whh, HH, j0 + 3 * HH, k);
    ai = wmma_bf16(a, b0, ai);
    af = wmma_bf16(a, b1, af);
    ag = wmma_bf16(a, b2, ag);
    ao = wmma_bf16(a, b3, ao);
  }
  int nl = j0 + (lane & 15);
  int mb = m0 + ((lane >> 4) << 3);
#pragma unroll
  for (int r = 0; r < 8; ++r) {
    int m = mb + r;
    float gI = ai[r] + gi[m * 4 * HH + 0 * HH + nl] + bih[0 * HH + nl] + bhh[0 * HH + nl];
    float gF = af[r] + gi[m * 4 * HH + 1 * HH + nl] + bih[1 * HH + nl] + bhh[1 * HH + nl];
    float gG = ag[r] + gi[m * 4 * HH + 2 * HH + nl] + bih[2 * HH + nl] + bhh[2 * HH + nl];
    float gO = ao[r] + gi[m * 4 * HH + 3 * HH + nl] + bih[3 * HH + nl] + bhh[3 * HH + nl];
    float iv = sigm(gI), fv = sigm(gF), gv = tanhf(gG), ov = sigm(gO);
    float cn = fv * c[m * HH + nl] + iv * gv;
    c[m * HH + nl] = cn;
    hOut[m * DH + nl] = to_bf16(ov * tanhf(cn));   // encOut slot, stride DH
  }
}

// ---------------------------------------------------------------------------
// Decoder recurrent step: gates = gx[t] + ctx@WihC^T + h@Whh^T + b; LSTM cell.
// Writes h into hNext and the h-half of the Z row for the batched logits GEMM.
// grid 32 x 128 threads (128 waves: jt 0..63 x mt 0..1).
// ---------------------------------------------------------------------------
__global__ void k_dec_step(const __bf16* __restrict__ hPrev,   // [B,DH]
                           const __bf16* __restrict__ ctx,     // z slot + DH, ld=2DH
                           const __bf16* __restrict__ WihC,    // dec_Wih cols E.., ld=E+DH
                           const __bf16* __restrict__ Whh,     // [4DH,DH]
                           const float* __restrict__ gx,       // [B,4DH]
                           const float* __restrict__ bih, const float* __restrict__ bhh,
                           float* __restrict__ c,              // [B,DH]
                           __bf16* __restrict__ hNext,         // [B,DH]
                           __bf16* __restrict__ zH) {          // z slot h-half, ld=2DH
  int wave = (blockIdx.x * blockDim.x + threadIdx.x) >> 5;     // 0..127
  int lane = threadIdx.x & 31;
  int jt = wave >> 1, mt = wave & 1;
  int j0 = jt * 16, m0 = mt * 16;
  v8f ai = {}, af = {}, ag = {}, ao = {};
  const int LW = EE + DH;                                      // 1280
  for (int k = 0; k < DH; k += 32) {
    v16bf a = load_afrag(ctx, 2 * DH, m0, k);
    ai = wmma_bf16(a, load_bfrag(WihC, LW, j0 + 0 * DH, k), ai);
    af = wmma_bf16(a, load_bfrag(WihC, LW, j0 + 1 * DH, k), af);
    ag = wmma_bf16(a, load_bfrag(WihC, LW, j0 + 2 * DH, k), ag);
    ao = wmma_bf16(a, load_bfrag(WihC, LW, j0 + 3 * DH, k), ao);
    v16bf a2 = load_afrag(hPrev, DH, m0, k);
    ai = wmma_bf16(a2, load_bfrag(Whh, DH, j0 + 0 * DH, k), ai);
    af = wmma_bf16(a2, load_bfrag(Whh, DH, j0 + 1 * DH, k), af);
    ag = wmma_bf16(a2, load_bfrag(Whh, DH, j0 + 2 * DH, k), ag);
    ao = wmma_bf16(a2, load_bfrag(Whh, DH, j0 + 3 * DH, k), ao);
  }
  int nl = j0 + (lane & 15);
  int mb = m0 + ((lane >> 4) << 3);
#pragma unroll
  for (int r = 0; r < 8; ++r) {
    int m = mb + r;
    float gI = ai[r] + gx[m * 4 * DH + 0 * DH + nl] + bih[0 * DH + nl] + bhh[0 * DH + nl];
    float gF = af[r] + gx[m * 4 * DH + 1 * DH + nl] + bih[1 * DH + nl] + bhh[1 * DH + nl];
    float gG = ag[r] + gx[m * 4 * DH + 2 * DH + nl] + bih[2 * DH + nl] + bhh[2 * DH + nl];
    float gO = ao[r] + gx[m * 4 * DH + 3 * DH + nl] + bih[3 * DH + nl] + bhh[3 * DH + nl];
    float iv = sigm(gI), fv = sigm(gF), gv = tanhf(gG), ov = sigm(gO);
    float cn = fv * c[m * DH + nl] + iv * gv;
    c[m * DH + nl] = cn;
    float hn = ov * tanhf(cn);
    hNext[m * DH + nl] = to_bf16(hn);
    zH[m * 2 * DH + nl] = to_bf16(hn);
  }
}

// ---------------------------------------------------------------------------
// Attention scores + softmax + context for one timestep. Block per batch row,
// 64 threads (one per source position).
// ---------------------------------------------------------------------------
__global__ void k_attn(const float* __restrict__ dproj,    // [B,A]
                       const float* __restrict__ encProj,  // [S,B,A]
                       const float* __restrict__ v_att, const float* __restrict__ bv,
                       const __bf16* __restrict__ encOut,  // [S,B,DH] bf16
                       __bf16* __restrict__ zCtx) {        // z slot + DH, ld=2DH
  __shared__ float sdp[AA];
  __shared__ float se[SS];
  int b = blockIdx.x;
  int tid = threadIdx.x;          // = s
  sdp[tid] = dproj[b * AA + tid];
  sdp[tid + 64] = dproj[b * AA + 64 + tid];
  __syncthreads();
  const float* ep = encProj + ((long long)tid * BB + b) * AA;
  float sc = 0.0f;
  for (int a = 0; a < AA; ++a) sc += tanhf(sdp[a] + ep[a]) * v_att[a];
  sc += bv[0];
  se[tid] = sc;
  __syncthreads();
  float mx = -1e30f;
  for (int s = 0; s < SS; ++s) mx = fmaxf(mx, se[s]);
  float e = __expf(sc - mx);
  __syncthreads();
  se[tid] = e;
  __syncthreads();
  float sum = 0.0f;
  for (int s = 0; s < SS; ++s) sum += se[s];
  float inv = 1.0f / sum;
  for (int j = tid; j < DH; j += 64) {
    float acc = 0.0f;
    for (int s = 0; s < SS; ++s)
      acc += se[s] * bf2f(encOut[((long long)s * BB + b) * DH + j]);
    zCtx[b * 2 * DH + j] = to_bf16(acc * inv);
  }
}

// Build decoder initial state h0=[hT_f,hT_b], c0=[cT_f,cT_b].
__global__ void k_dec_init(const __bf16* __restrict__ encOut,
                           const float* __restrict__ cF, const float* __restrict__ cB,
                           __bf16* __restrict__ h0, float* __restrict__ c0) {
  int idx = blockIdx.x * blockDim.x + threadIdx.x;
  if (idx >= BB * DH) return;
  int b = idx >> 10, j = idx & (DH - 1);
  __bf16 hv = (j < HH) ? encOut[((long long)(SS - 1) * BB + b) * DH + j]
                       : encOut[(long long)b * DH + j];
  h0[idx] = hv;
  c0[idx] = (j < HH) ? cF[b * HH + j] : cB[b * HH + (j - HH)];
}

// ---------------------------------------------------------------------------
// Host launcher
// ---------------------------------------------------------------------------
extern "C" void kernel_launch(void* const* d_in, const int* in_sizes, int n_in,
                              void* d_out, int out_size, void* d_ws, size_t ws_size,
                              hipStream_t stream) {
  (void)in_sizes; (void)n_in; (void)out_size; (void)ws_size;
  const int* src = (const int*)d_in[0];
  const int* tgt = (const int*)d_in[1];
  const float* enc_emb = (const float*)d_in[2];
  const float* eWih_f = (const float*)d_in[3];
  const float* eWhh_f = (const float*)d_in[4];
  const float* ebih_f = (const float*)d_in[5];
  const float* ebhh_f = (const float*)d_in[6];
  const float* eWih_b = (const float*)d_in[7];
  const float* eWhh_b = (const float*)d_in[8];
  const float* ebih_b = (const float*)d_in[9];
  const float* ebhh_b = (const float*)d_in[10];
  const float* dec_emb = (const float*)d_in[11];
  const float* Wd = (const float*)d_in[12];
  const float* bd = (const float*)d_in[13];
  const float* We = (const float*)d_in[14];
  const float* be = (const float*)d_in[15];
  const float* v_att = (const float*)d_in[16];
  const float* bv = (const float*)d_in[17];
  const float* dWih = (const float*)d_in[18];
  const float* dWhh = (const float*)d_in[19];
  const float* dbih = (const float*)d_in[20];
  const float* dbhh = (const float*)d_in[21];
  const float* Wo = (const float*)d_in[22];
  const float* bo = (const float*)d_in[23];
  float* out = (float*)d_out;

  // ---- workspace carve-up ----
  char* base = (char*)d_ws;
  size_t off = 0;
  auto alloc = [&](size_t bytes) -> void* {
    void* p = base + off;
    off = (off + bytes + 255) & ~(size_t)255;
    return p;
  };
  __bf16* wWihF = (__bf16*)alloc((size_t)4 * HH * EE * 2);
  __bf16* wWhhF = (__bf16*)alloc((size_t)4 * HH * HH * 2);
  __bf16* wWihB = (__bf16*)alloc((size_t)4 * HH * EE * 2);
  __bf16* wWhhB = (__bf16*)alloc((size_t)4 * HH * HH * 2);
  __bf16* wDWih = (__bf16*)alloc((size_t)4 * DH * (EE + DH) * 2);
  __bf16* wDWhh = (__bf16*)alloc((size_t)4 * DH * DH * 2);
  __bf16* wWd   = (__bf16*)alloc((size_t)AA * DH * 2);
  __bf16* wWe   = (__bf16*)alloc((size_t)AA * DH * 2);
  __bf16* wWo   = (__bf16*)alloc((size_t)VT * 2 * DH * 2);
  __bf16* xEnc  = (__bf16*)alloc((size_t)SS * BB * EE * 2);
  __bf16* xDec  = (__bf16*)alloc((size_t)TT * BB * EE * 2);
  float*  giF   = (float*)alloc((size_t)SS * BB * 4 * HH * 4);
  float*  giB   = (float*)alloc((size_t)SS * BB * 4 * HH * 4);
  float*  gxDec = (float*)alloc((size_t)TT * BB * 4 * DH * 4);
  __bf16* encOut = (__bf16*)alloc((size_t)SS * BB * DH * 2);
  float*  encProj = (float*)alloc((size_t)SS * BB * AA * 4);
  float*  cEncF = (float*)alloc((size_t)BB * HH * 4);
  float*  cEncB = (float*)alloc((size_t)BB * HH * 4);
  __bf16* zerosH = (__bf16*)alloc((size_t)BB * HH * 2);
  __bf16* hDec0 = (__bf16*)alloc((size_t)BB * DH * 2);
  __bf16* hDec1 = (__bf16*)alloc((size_t)BB * DH * 2);
  float*  cDec  = (float*)alloc((size_t)BB * DH * 4);
  __bf16* Z     = (__bf16*)alloc((size_t)TT * BB * 2 * DH * 2);  // [t*B+b, 2DH]
  float*  dproj = (float*)alloc((size_t)BB * AA * 4);

  auto cvt = [&](const float* s, __bf16* d, long long n) {
    int blocks = (int)((n + 255) / 256);
    if (blocks > 16384) blocks = 16384;
    k_cvt<<<blocks, 256, 0, stream>>>(s, d, n);
  };
  auto gblk = [](int tiles) { return (tiles + 3) / 4; };  // 4 waves / block

  // ---- phase 0: state init + weight conversion + embedding gathers ----
  k_zero_f32<<<(BB * HH + 255) / 256, 256, 0, stream>>>(cEncF, BB * HH);
  k_zero_f32<<<(BB * HH + 255) / 256, 256, 0, stream>>>(cEncB, BB * HH);
  k_zero_bf16<<<(BB * HH + 255) / 256, 256, 0, stream>>>(zerosH, BB * HH);
  cvt(eWih_f, wWihF, (long long)4 * HH * EE);
  cvt(eWhh_f, wWhhF, (long long)4 * HH * HH);
  cvt(eWih_b, wWihB, (long long)4 * HH * EE);
  cvt(eWhh_b, wWhhB, (long long)4 * HH * HH);
  cvt(dWih, wDWih, (long long)4 * DH * (EE + DH));
  cvt(dWhh, wDWhh, (long long)4 * DH * DH);
  cvt(Wd, wWd, (long long)AA * DH);
  cvt(We, wWe, (long long)AA * DH);
  cvt(Wo, wWo, (long long)VT * 2 * DH);
  k_embed<<<SS * BB, EE, 0, stream>>>(src, enc_emb, xEnc, SS);
  k_embed<<<TT * BB, EE, 0, stream>>>(tgt, dec_emb, xDec, TT);

  // ---- phase 1: hoisted input projections (big WMMA GEMMs) ----
  // giF/giB = xEnc @ Wih^T   [2048 x 2048, K=256]
  k_gemm<<<gblk((SS * BB / 16) * (4 * HH / 16)), 128, 0, stream>>>(
      xEnc, EE, wWihF, EE, nullptr, giF, 4 * HH, SS * BB, 4 * HH, EE);
  k_gemm<<<gblk((SS * BB / 16) * (4 * HH / 16)), 128, 0, stream>>>(
      xEnc, EE, wWihB, EE, nullptr, giB, 4 * HH, SS * BB, 4 * HH, EE);
  // gxDec = xDec @ dec_Wih[:, :E]^T   [2048 x 4096, K=256]
  k_gemm<<<gblk((TT * BB / 16) * (4 * DH / 16)), 128, 0, stream>>>(
      xDec, EE, wDWih, EE + DH, nullptr, gxDec, 4 * DH, TT * BB, 4 * DH, EE);

  // ---- phase 2: encoder recurrence (both directions fused per iteration) ----
  for (int it = 0; it < SS; ++it) {
    int sF = it, sB = SS - 1 - it;
    const __bf16* hPF = (it == 0) ? zerosH : encOut + (size_t)(sF - 1) * BB * DH;
    int ldF = (it == 0) ? HH : DH;
    const __bf16* hPB = (it == 0) ? zerosH : encOut + (size_t)(sB + 1) * BB * DH + HH;
    int ldB = (it == 0) ? HH : DH;
    k_enc_step_bi<<<32, 128, 0, stream>>>(
        hPF, ldF, hPB, ldB, wWhhF, wWhhB,
        giF + (size_t)sF * BB * 4 * HH, giB + (size_t)sB * BB * 4 * HH,
        ebih_f, ebhh_f, ebih_b, ebhh_b, cEncF, cEncB,
        encOut + (size_t)sF * BB * DH, encOut + (size_t)sB * BB * DH + HH);
  }

  // ---- phase 3: attention precompute + decoder init ----
  // encProj = encOut @ We^T + be   [2048 x 128, K=1024]
  k_gemm<<<gblk((SS * BB / 16) * (AA / 16)), 128, 0, stream>>>(
      encOut, DH, wWe, DH, be, encProj, AA, SS * BB, AA, DH);
  k_dec_init<<<(BB * DH + 255) / 256, 256, 0, stream>>>(encOut, cEncF, cEncB, hDec0, cDec);

  // ---- phase 4: decoder recurrence ----
  for (int t = 0; t < TT; ++t) {
    __bf16* hCur = (t & 1) ? hDec1 : hDec0;
    __bf16* hNxt = (t & 1) ? hDec0 : hDec1;
    __bf16* zSlot = Z + (size_t)t * BB * 2 * DH;
    // dproj = h @ Wd^T + bd   [32 x 128, K=1024]
    k_gemm<<<gblk((BB / 16) * (AA / 16)), 128, 0, stream>>>(
        hCur, DH, wWd, DH, bd, dproj, AA, BB, AA, DH);
    // scores/softmax/context -> ctx half of Z row
    k_attn<<<BB, SS, 0, stream>>>(dproj, encProj, v_att, bv, encOut, zSlot + DH);
    // recurrent gates + cell -> h half of Z row
    k_dec_step<<<32, 128, 0, stream>>>(
        hCur, zSlot + DH, wDWih + EE, wDWhh,
        gxDec + (size_t)t * BB * 4 * DH, dbih, dbhh, cDec, hNxt, zSlot);
  }

  // ---- phase 5: batched logits, Wo streamed once; 64x64 macro-tile/wave ----
  k_gemm_logits<<<gblk(((TT * BB / 16) / 4) * ((VT / 16) / 4)), 128, 0, stream>>>(
      Z, wWo, bo, out);
}
